// Net_47751446397014
// MI455X (gfx1250) — compile-verified, matrix operand order checked by reading, not run on hardware
//
#include <hip/hip_runtime.h>
#include <hip/hip_bf16.h>
#include <math.h>

#define NN 6890
#define EE 41340
#define KS_ 5
#define KK 25          // KS*KS
#define MPAD 6896      // 431 * 16

typedef __bf16 bf16_t;
typedef __attribute__((ext_vector_type(16))) __bf16 v16bf;
typedef __attribute__((ext_vector_type(8)))  __bf16 v8bf;
typedef __attribute__((ext_vector_type(8)))  float  v8f;

__device__ __forceinline__ float elu1(float v) { return v > 0.f ? v : (expf(v) - 1.f); }

union Frag { v16bf v; struct { v8bf lo, hi; } h; };

// ---------------------------------------------------------------------------
// WMMA GEMM, branch-free inner loop.
//   A  : bf16 [Mpad, Kapad] row-major, Kapad % 32 == 0, rows >= M zero-filled
//   Bp : bf16 fragment-packed tiles: element ((kt*Tnp + nt)*32 + lane)*16 + i
//   One wave32 computes a 16x64 tile (4 accumulators, A-fragment reuse).
//   All 10 b128 loads of a K-step are issued before the first WMMA so waits
//   can be partial; no values besides accumulators live across iterations
//   (keeps VGPR pressure low enough to avoid spills).
// ---------------------------------------------------------------------------
__global__ void gemm_wmma_bf16(const bf16_t* __restrict__ A,
                               const bf16_t* __restrict__ Bp,
                               const float* __restrict__ bias,
                               float* __restrict__ Cf, int ldf,
                               bf16_t* __restrict__ Cbf, int ldb16,
                               int M, int Kapad, int Nc, int Tnp, int act) {
  const int lane  = threadIdx.x;            // 0..31 (wave32)
  const int m0    = blockIdx.x * 16;
  const int nt0   = blockIdx.y * 4;
  const int nl    = lane & 15;
  const int khalf = (lane >> 4) * 8;        // lanes 16..31 hold K+8 / K+24 runs
  const bf16_t* Arow = A + (size_t)(m0 + nl) * Kapad;
  const int Tk = Kapad >> 5;

  v8f c0 = {}, c1 = {}, c2 = {}, c3 = {};

  for (int kt = 0; kt < Tk; ++kt) {
    if (kt + 1 < Tk)  // global_prefetch_b8
      __builtin_prefetch(Arow + (size_t)(kt + 1) * 32 + khalf, 0, 1);

    // ---- issue all fragment loads for this K-step up front ----
    Frag a, b0, b1, b2, b3;
    const bf16_t* ap = Arow + (size_t)kt * 32 + khalf;
    a.h.lo = *(const v8bf*)(ap);        // K = kt*32 + khalf + 0..7
    a.h.hi = *(const v8bf*)(ap + 16);   // K = kt*32 + 16 + khalf + 0..7

    const bf16_t* bt = Bp + (((size_t)kt * Tnp + nt0) * 32 + lane) * 16;
    b0.h.lo = *(const v8bf*)(bt);        b0.h.hi = *(const v8bf*)(bt + 8);
    b1.h.lo = *(const v8bf*)(bt + 512);  b1.h.hi = *(const v8bf*)(bt + 520);
    b2.h.lo = *(const v8bf*)(bt + 1024); b2.h.hi = *(const v8bf*)(bt + 1032);
    b3.h.lo = *(const v8bf*)(bt + 1536); b3.h.hi = *(const v8bf*)(bt + 1544);

    // ---- 4 WMMAs sharing the A fragment ----
    c0 = __builtin_amdgcn_wmma_f32_16x16x32_bf16(false, a.v, false, b0.v, (short)0, c0, false, false);
    c1 = __builtin_amdgcn_wmma_f32_16x16x32_bf16(false, a.v, false, b1.v, (short)0, c1, false, false);
    c2 = __builtin_amdgcn_wmma_f32_16x16x32_bf16(false, a.v, false, b2.v, (short)0, c2, false, false);
    c3 = __builtin_amdgcn_wmma_f32_16x16x32_bf16(false, a.v, false, b3.v, (short)0, c3, false, false);
  }

  v8f cc[4] = {c0, c1, c2, c3};
#pragma unroll
  for (int t = 0; t < 4; ++t) {
    const int n = (nt0 + t) * 16 + nl;
    if (n >= Nc) continue;
    const float bia = bias ? bias[n] : 0.f;
#pragma unroll
    for (int r = 0; r < 8; ++r) {           // C VGPR r: M = r (+8 for hi lanes)
      const int mr = m0 + r + ((lane >> 4) << 3);
      if (mr >= M) continue;
      float v = cc[t][r] + bia;
      if (act) v = elu1(v);
      if (Cf)  Cf [(size_t)mr * ldf   + n] = v;
      if (Cbf) Cbf[(size_t)mr * ldb16 + n] = (bf16_t)v;
    }
  }
}

// ---------------------------------------------------------------------------
// B packing: f32 [Ka,Nc] -> fragment-ready bf16 tiles (zero padded)
// ---------------------------------------------------------------------------
__device__ __forceinline__ void pack_decode(int id, int Tnp, int& k, int& n, int& outid) {
  outid = id;
  const int idx  = id & 15;
  const int lane = (id >> 4) & 31;
  const int tile = id >> 9;
  const int nt = tile % Tnp, kt = tile / Tnp;
  const int khalf = (lane >> 4) * 8;
  const int kl = (idx < 8) ? (khalf + idx) : (16 + khalf + (idx - 8));
  k = kt * 32 + kl;
  n = nt * 16 + (lane & 15);
}

__global__ void pack_B_plain(const float* __restrict__ B, bf16_t* __restrict__ out,
                             int Ka, int Nc, int Tk, int Tnp) {
  int id = blockIdx.x * blockDim.x + threadIdx.x;
  if (id >= Tk * Tnp * 512) return;
  int k, n, oid;
  pack_decode(id, Tnp, k, n, oid);
  float v = (k < Ka && n < Nc) ? B[(size_t)k * Nc + n] : 0.f;
  out[oid] = (bf16_t)v;
}

// spline weight [KK, Fin, Fout] viewed as B[Fin, KK*Fout]
__global__ void pack_B_spline(const float* __restrict__ w, bf16_t* __restrict__ out,
                              int Fin, int Fout, int Tk, int Tnp) {
  int id = blockIdx.x * blockDim.x + threadIdx.x;
  if (id >= Tk * Tnp * 512) return;
  int k, n, oid;
  pack_decode(id, Tnp, k, n, oid);
  float v = 0.f;
  if (k < Fin && n < KK * Fout) {
    int kk = n / Fout, o = n % Fout;
    v = w[((size_t)kk * Fin + k) * Fout + o];
  }
  out[oid] = (bf16_t)v;
}

// ---------------------------------------------------------------------------
// Small helper kernels
// ---------------------------------------------------------------------------
__global__ void cast_f32_bf16(const float* __restrict__ in, bf16_t* __restrict__ out, int n) {
  int i = blockIdx.x * blockDim.x + threadIdx.x;
  if (i < n) out[i] = (bf16_t)in[i];
}

__global__ void fill_bf16(bf16_t* p, int n) {
  int i = blockIdx.x * blockDim.x + threadIdx.x;
  if (i < n) p[i] = (bf16_t)0.f;
}

__global__ void fill_f32(float* p, float v, int n) {
  int i = blockIdx.x * blockDim.x + threadIdx.x;
  if (i < n) p[i] = v;
}

__global__ void deg_kernel(const int* __restrict__ ei, float* __restrict__ deg) {
  int e = blockIdx.x * blockDim.x + threadIdx.x;
  if (e < EE) atomicAdd(&deg[ei[e]], 1.0f);
}

// per-edge degree-1 open B-spline blend + scatter-add into agg
__global__ void spline_edge(const int* __restrict__ ei, const float* __restrict__ pseudo,
                            const float* __restrict__ xw, float* __restrict__ agg,
                            int Fout, int KFout) {
  const int e = blockIdx.x;
  const int f = threadIdx.x;
  if (f >= Fout) return;
  const int row = ei[e];
  const int col = ei[EE + e];
  float px = pseudo[2 * e + 0] * (float)(KS_ - 1);
  float py = pseudo[2 * e + 1] * (float)(KS_ - 1);
  float i0x = fminf(fmaxf(floorf(px), 0.f), (float)(KS_ - 2));
  float i0y = fminf(fmaxf(floorf(py), 0.f), (float)(KS_ - 2));
  float fx = px - i0x, fy = py - i0y;
  float bx[2] = {1.f - fx, fx};
  float by[2] = {1.f - fy, fy};
  int ix = (int)i0x, iy = (int)i0y;
  const float* src = xw + (size_t)col * KFout;
  float acc = 0.f;
#pragma unroll
  for (int oy = 0; oy < 2; ++oy)
#pragma unroll
    for (int ox = 0; ox < 2; ++ox) {
      int kk = (ix + ox) + KS_ * (iy + oy);
      acc += bx[ox] * by[oy] * src[kk * Fout + f];
    }
  atomicAdd(&agg[(size_t)row * Fout + f], acc);
}

// h = elu(agg/max(deg,1) + xroot)  (bias folded into xroot GEMM)
__global__ void finish_node(const float* __restrict__ agg, const float* __restrict__ deg,
                            const float* __restrict__ xroot, bf16_t* __restrict__ hb,
                            int Fout) {
  int i = blockIdx.x * blockDim.x + threadIdx.x;
  if (i >= NN * Fout) return;
  int nrow = i / Fout;
  float d = fmaxf(deg[nrow], 1.0f);
  hb[i] = (bf16_t)elu1(agg[i] / d + xroot[i]);
}

// in-place row log-softmax, one block per row
__global__ void log_softmax_rows(float* __restrict__ out, int cols) {
  const int r = blockIdx.x;
  float* rowp = out + (size_t)r * cols;
  __shared__ float red[256];
  const int t = threadIdx.x;

  float m = -3.4e38f;
  for (int c = t; c < cols; c += blockDim.x) m = fmaxf(m, rowp[c]);
  red[t] = m; __syncthreads();
  for (int s = blockDim.x / 2; s > 0; s >>= 1) {
    if (t < s) red[t] = fmaxf(red[t], red[t + s]);
    __syncthreads();
  }
  m = red[0]; __syncthreads();

  float sum = 0.f;
  for (int c = t; c < cols; c += blockDim.x) sum += expf(rowp[c] - m);
  red[t] = sum; __syncthreads();
  for (int s = blockDim.x / 2; s > 0; s >>= 1) {
    if (t < s) red[t] += red[t + s];
    __syncthreads();
  }
  const float l = logf(red[0]);
  __syncthreads();

  for (int c = t; c < cols; c += blockDim.x) rowp[c] = rowp[c] - m - l;
}

// ---------------------------------------------------------------------------
extern "C" void kernel_launch(void* const* d_in, const int* in_sizes, int n_in,
                              void* d_out, int out_size, void* d_ws, size_t ws_size,
                              hipStream_t stream) {
  (void)in_sizes; (void)n_in; (void)out_size; (void)ws_size;
  const float* x      = (const float*)d_in[0];
  const int*   ei     = (const int*)  d_in[1];
  const float* pseudo = (const float*)d_in[2];
  const float* W_lin  = (const float*)d_in[3];
  const float* b_lin  = (const float*)d_in[4];
  const float* w1     = (const float*)d_in[5];
  const float* r1     = (const float*)d_in[6];
  const float* bi1    = (const float*)d_in[7];
  const float* w2     = (const float*)d_in[8];
  const float* r2     = (const float*)d_in[9];
  const float* bi2    = (const float*)d_in[10];
  const float* w3     = (const float*)d_in[11];
  const float* r3     = (const float*)d_in[12];
  const float* bi3    = (const float*)d_in[13];
  const float* W1     = (const float*)d_in[14];
  const float* b1     = (const float*)d_in[15];
  const float* W2     = (const float*)d_in[16];
  const float* b2     = (const float*)d_in[17];

  char* ws = (char*)d_ws;
  size_t off = 0;
  auto alloc = [&](size_t bytes) -> void* {
    void* p = ws + off;
    off = (off + bytes + 255) & ~(size_t)255;
    return p;
  };
  auto tnp = [](int Nc) { return (((Nc + 15) / 16) + 3) & ~3; };  // N tiles, mult of 4

  // packed-B tile counts
  const int TnWlin = tnp(16),       TkWlin = 544 / 32;
  const int TnW1t  = tnp(KK * 32),  TkW1t  = 1;    // Fin 16 padded to 32
  const int TnR1   = tnp(32),       TkR1   = 1;
  const int TnW2t  = tnp(KK * 64),  TkW2t  = 1;
  const int TnR2   = tnp(64),       TkR2   = 1;
  const int TnW3t  = tnp(KK * 128), TkW3t  = 2;
  const int TnR3   = tnp(128),      TkR3   = 2;
  const int TnWd1  = tnp(256),      TkWd1  = 4;
  const int TnWd2  = tnp(NN),       TkWd2  = 8;

  bf16_t* x_bf   = (bf16_t*)alloc((size_t)MPAD * 544 * 2);
  bf16_t* wlinP  = (bf16_t*)alloc((size_t)TkWlin * TnWlin * 512 * 2);
  bf16_t* w1tP   = (bf16_t*)alloc((size_t)TkW1t  * TnW1t  * 512 * 2);
  bf16_t* r1P    = (bf16_t*)alloc((size_t)TkR1   * TnR1   * 512 * 2);
  bf16_t* w2tP   = (bf16_t*)alloc((size_t)TkW2t  * TnW2t  * 512 * 2);
  bf16_t* r2P    = (bf16_t*)alloc((size_t)TkR2   * TnR2   * 512 * 2);
  bf16_t* w3tP   = (bf16_t*)alloc((size_t)TkW3t  * TnW3t  * 512 * 2);
  bf16_t* r3P    = (bf16_t*)alloc((size_t)TkR3   * TnR3   * 512 * 2);
  bf16_t* W1P    = (bf16_t*)alloc((size_t)TkWd1  * TnWd1  * 512 * 2);
  bf16_t* W2P    = (bf16_t*)alloc((size_t)TkWd2  * TnWd2  * 512 * 2);
  bf16_t* h0b    = (bf16_t*)alloc((size_t)MPAD * 32 * 2);   // Nc=16, Kpad(next)=32
  bf16_t* h1b    = (bf16_t*)alloc((size_t)MPAD * 32 * 2);
  bf16_t* h2b    = (bf16_t*)alloc((size_t)MPAD * 64 * 2);
  bf16_t* h3b    = (bf16_t*)alloc((size_t)MPAD * 128 * 2);
  bf16_t* h4b    = (bf16_t*)alloc((size_t)MPAD * 256 * 2);
  float*  xw     = (float*) alloc((size_t)NN * (KK * 128) * 4);  // reused
  float*  xroot  = (float*) alloc((size_t)NN * 128 * 4);         // reused
  float*  agg    = (float*) alloc((size_t)NN * 128 * 4);         // reused
  float*  deg    = (float*) alloc((size_t)NN * 4);

  auto gemm = [&](const bf16_t* A, const bf16_t* Bp, const float* bias,
                  float* Cf, int ldf, bf16_t* Cbf, int ldb,
                  int M, int Kapad, int Nc, int Tnp_, int act) {
    dim3 g(MPAD / 16, Tnp_ / 4);
    gemm_wmma_bf16<<<g, dim3(32), 0, stream>>>(A, Bp, bias, Cf, ldf, Cbf, ldb,
                                               M, Kapad, Nc, Tnp_, act);
  };
  auto zero_bf = [&](bf16_t* p, size_t n) {
    fill_bf16<<<((int)n + 255) / 256, 256, 0, stream>>>(p, (int)n);
  };
  auto packP = [&](const float* B, bf16_t* out, int Ka, int Nc, int Tk, int Tnp_) {
    int tot = Tk * Tnp_ * 512;
    pack_B_plain<<<(tot + 255) / 256, 256, 0, stream>>>(B, out, Ka, Nc, Tk, Tnp_);
  };
  auto packS = [&](const float* w, bf16_t* out, int Fin, int Fout, int Tk, int Tnp_) {
    int tot = Tk * Tnp_ * 512;
    pack_B_spline<<<(tot + 255) / 256, 256, 0, stream>>>(w, out, Fin, Fout, Tk, Tnp_);
  };

  // ---- zero padded activation buffers (pad rows / pad cols must be 0) ----
  zero_bf(x_bf, (size_t)MPAD * 544);
  zero_bf(h0b, (size_t)MPAD * 32);
  zero_bf(h1b, (size_t)MPAD * 32);
  zero_bf(h2b, (size_t)MPAD * 64);
  zero_bf(h3b, (size_t)MPAD * 128);
  zero_bf(h4b, (size_t)MPAD * 256);

  // ---- input cast + weight packing ----
  cast_f32_bf16<<<(NN * 544 + 255) / 256, 256, 0, stream>>>(x, x_bf, NN * 544);
  packP(W_lin, wlinP, 544, 16, TkWlin, TnWlin);
  packS(w1, w1tP, 16, 32, TkW1t, TnW1t);
  packP(r1, r1P, 16, 32, TkR1, TnR1);
  packS(w2, w2tP, 32, 64, TkW2t, TnW2t);
  packP(r2, r2P, 32, 64, TkR2, TnR2);
  packS(w3, w3tP, 64, 128, TkW3t, TnW3t);
  packP(r3, r3P, 64, 128, TkR3, TnR3);
  packP(W1, W1P, 128, 256, TkWd1, TnWd1);
  packP(W2, W2P, 256, NN, TkWd2, TnWd2);

  // ---- degrees (shared by all conv layers) ----
  fill_f32<<<(NN + 255) / 256, 256, 0, stream>>>(deg, 0.f, NN);
  deg_kernel<<<(EE + 255) / 256, 256, 0, stream>>>(ei, deg);

  // ---- input linear + ELU -> h0 (stored with K padded to 32) ----
  gemm(x_bf, wlinP, b_lin, nullptr, 0, h0b, 32, NN, 544, 16, TnWlin, 1);

  // ---- spline conv layers ----
  struct Layer {
    const bf16_t *wt, *rb, *hin; const float* bi; bf16_t* hout;
    int Kapad, Fout, TnW, TnR;
  };
  Layer layers[3] = {
    { w1tP, r1P, h0b, bi1, h1b, 32, 32,  TnW1t, TnR1 },
    { w2tP, r2P, h1b, bi2, h2b, 32, 64,  TnW2t, TnR2 },
    { w3tP, r3P, h2b, bi3, h3b, 64, 128, TnW3t, TnR3 },
  };
  for (int L = 0; L < 3; ++L) {
    const Layer& ly = layers[L];
    const int KF = KK * ly.Fout;
    gemm(ly.hin, ly.wt, nullptr, xw, KF, nullptr, 0, NN, ly.Kapad, KF, ly.TnW, 0);
    gemm(ly.hin, ly.rb, ly.bi, xroot, ly.Fout, nullptr, 0, NN, ly.Kapad, ly.Fout, ly.TnR, 0);
    fill_f32<<<(NN * ly.Fout + 255) / 256, 256, 0, stream>>>(agg, 0.f, NN * ly.Fout);
    spline_edge<<<EE, ly.Fout, 0, stream>>>(ei, pseudo, xw, agg, ly.Fout, KF);
    finish_node<<<(NN * ly.Fout + 255) / 256, 256, 0, stream>>>(agg, deg, xroot,
                                                                ly.hout, ly.Fout);
  }

  // ---- head: MLP + classifier into d_out, then in-place log-softmax ----
  gemm(h3b, W1P, b1, nullptr, 0, h4b, 256, NN, 128, 256, TnWd1, 1);
  gemm(h4b, W2P, b2, (float*)d_out, NN, nullptr, 0, NN, 256, NN, TnWd2, 0);
  log_softmax_rows<<<NN, 256, 0, stream>>>((float*)d_out, NN);
}
